// Model_78211354460498
// MI455X (gfx1250) — matrix operand order repaired and checked by reading, not compiled, not run on hardware
//
#include <hip/hip_runtime.h>
#include <hip/hip_bf16.h>
#include <cstdint>
#include <cstddef>

// ---------------------------------------------------------------------------
// Types for CDNA5 WMMA (gfx1250, wave32): D(16x16 f32) = A(16x32 bf16) x B(32x16 bf16) + C
// ---------------------------------------------------------------------------
typedef __attribute__((ext_vector_type(16))) __bf16 v16bf;
typedef __attribute__((ext_vector_type(8)))  float  v8f;

__device__ __forceinline__ unsigned short f2bf(float f) {
    unsigned u = __float_as_uint(f);
    u += 0x7FFFu + ((u >> 16) & 1u);          // round-to-nearest-even
    return (unsigned short)(u >> 16);
}
__device__ __forceinline__ float bf2f(unsigned short h) {
    return __uint_as_float(((unsigned)h) << 16);
}

// ---------------------------------------------------------------------------
// Weight repack: f32 OIHW -> bf16 packed directly in the WMMA B-fragment layout
//   P[kb][nt][lane][j]  (kb = K/32 blocks, nt = Cout/16 tiles, lane = 0..31, j = 0..15)
//   fragment semantics: value = W[k = kb*32 + (j>>3)*16 + (lane>>4)*8 + (j&7)][n = nt*16 + (lane&15)]
//   K zero-padded to a multiple of 32 (handles the K=147 stem).
// ---------------------------------------------------------------------------
__global__ void k_prep_w(const float* __restrict__ w, unsigned short* __restrict__ out,
                         int O, int I, int KH, int KW, int Kpad) {
    long e = (long)blockIdx.x * 256 + threadIdx.x;
    long total = (long)Kpad * O;
    if (e >= total) return;
    int  j    = (int)(e & 15);
    long t    = e >> 4;
    int  lane = (int)(t & 31); t >>= 5;
    int  NT   = O >> 4;
    int  nt   = (int)(t % NT);
    int  kb   = (int)(t / NT);
    int  k    = kb * 32 + (j >> 3) * 16 + (lane >> 4) * 8 + (j & 7);
    int  n    = nt * 16 + (lane & 15);
    int  Kr   = KH * KW * I;
    unsigned short v = 0;
    if (k < Kr) {
        int ci = k % I;
        int tt = k / I;
        int kx = tt % KW;
        int ky = tt / KW;
        v = f2bf(w[(((long)n * I + ci) * KH + ky) * KW + kx]);
    }
    out[e] = v;
}

// BN fold: scale = g*rsqrt(v+eps), bias = b - m*scale
__global__ void k_prep_bn(const float* g, const float* b, const float* m, const float* v,
                          float* scale, float* bias, int C) {
    int c = blockIdx.x * 256 + threadIdx.x;
    if (c >= C) return;
    float inv = g[c] * rsqrtf(v[c] + 1e-5f);
    scale[c] = inv;
    bias[c]  = b[c] - m[c] * inv;
}

// Input: f32 NCHW -> bf16 NHWC
__global__ void k_x_to_nhwc(const float* __restrict__ x, unsigned short* __restrict__ out,
                            int B, int C, int H, int W) {
    long e = (long)blockIdx.x * 256 + threadIdx.x;
    long total = (long)B * C * H * W;
    if (e >= total) return;
    int c = (int)(e % C);
    long t = e / C;
    int xx = (int)(t % W); t /= W;
    int y  = (int)(t % H);
    int b  = (int)(t / H);
    out[e] = f2bf(x[(((long)b * C + c) * H + y) * W + xx]);
}

// 3x3 s2 p1 maxpool, bf16 NHWC
__global__ void k_maxpool(const unsigned short* __restrict__ in, unsigned short* __restrict__ out,
                          int B, int C, int H, int W, int Ho, int Wo) {
    long e = (long)blockIdx.x * 256 + threadIdx.x;
    long total = (long)B * Ho * Wo * C;
    if (e >= total) return;
    int c = (int)(e % C);
    long t = e / C;
    int ox = (int)(t % Wo); t /= Wo;
    int oy = (int)(t % Ho);
    int b  = (int)(t / Ho);
    float mx = -3.4e38f;
    for (int ky = 0; ky < 3; ky++) {
        int iy = oy * 2 - 1 + ky;
        if (iy < 0 || iy >= H) continue;
        for (int kx = 0; kx < 3; kx++) {
            int ix = ox * 2 - 1 + kx;
            if (ix < 0 || ix >= W) continue;
            float v = bf2f(in[(((long)b * H + iy) * W + ix) * C + c]);
            mx = fmaxf(mx, v);
        }
    }
    out[e] = f2bf(mx);
}

// ---------------------------------------------------------------------------
// Fused implicit-GEMM conv + BN + residual + ReLU with v_wmma_f32_16x16x32_bf16.
//   M = B*Ho*Wo pixels, N = Cout, K = KH*KW*Cin. Block tile 128x64, 8 wave32 (4Mx2N),
//   each wave owns a 32x32 tile (4 accumulators).
//   A: im2col gather -> LDS staged *in fragment order* -> per-lane ds_load_b128.
//      Cin%32==0 path uses contiguous 16B global loads and 16B LDS stores.
//   B: loaded straight from the pre-packed fragment-layout weights with
//      per-lane 32B global loads (no LDS round trip; reused via WGP$).
// ---------------------------------------------------------------------------
__global__ __launch_bounds__(256) void k_conv_wmma(
    const unsigned short* __restrict__ X,    // bf16 NHWC [B,H,W,Cin]
    const unsigned short* __restrict__ Wt,   // bf16 packed [Kpad/32][Cout/16][32][16]
    const float* __restrict__ scale,         // BN scale (nullptr = identity)
    const float* __restrict__ bias,
    const unsigned short* __restrict__ Res,  // bf16 NHWC residual (nullptr = none)
    unsigned short* __restrict__ Y,          // bf16 NHWC out (nullable)
    float* __restrict__ Yf,                  // f32 NHWC out (nullable)
    int Bn, int H, int W, int Cin, int Cout,
    int KH, int KW, int stride, int pad, int Ho, int Wo, int relu)
{
    __shared__ unsigned short aA[8][32][16]; // 8 M-subtiles of per-lane A fragments (8KB)

    const int M    = Bn * Ho * Wo;
    const int K    = KH * KW * Cin;
    const int HoWo = Ho * Wo;
    const int NT   = Cout >> 4;

    const int tid  = threadIdx.x;
    const int lane = tid & 31;
    const int wv   = tid >> 5;   // 0..7
    const int wm   = wv & 3;     // wave M tile 0..3
    const int wn   = wv >> 2;    // wave N tile 0..1
    const int lm   = lane & 15;
    const int lh   = lane >> 4;

    const int mBase  = blockIdx.x * 128;
    const int ntBase = blockIdx.y * 4;   // 4 x 16 = 64 output channels per block

    v8f acc[2][2];
#pragma unroll
    for (int a = 0; a < 2; a++)
#pragma unroll
        for (int bb = 0; bb < 2; bb++)
#pragma unroll
            for (int r = 0; r < 8; r++) acc[a][bb][r] = 0.0f;

    const bool fast = (Cin & 31) == 0;   // every conv except the Cin=3 stem

    for (int k0 = 0; k0 < K; k0 += 32) {
        const int kb = k0 >> 5;
        if (fast) {
            // 32-wide K chunk lies inside one (ky,kx) slice: contiguous channels.
            const int t   = k0 / Cin;          // uniform over block
            const int ci0 = k0 - t * Cin;
            const int kx  = t % KW;
            const int ky  = t / KW;
#pragma unroll
            for (int e = tid; e < 512; e += 256) {      // 128 rows x 4 chunks of 8 bf16
                int r = e >> 2, ch = e & 3;
                int m = mBase + r;
                uint4 v; v.x = v.y = v.z = v.w = 0u;
                if (m < M) {
                    int bb = m / HoWo;
                    int p  = m - bb * HoWo;
                    int oy = p / Wo;
                    int ox = p - oy * Wo;
                    int iy = oy * stride - pad + ky;
                    int ix = ox * stride - pad + kx;
                    if (iy >= 0 && iy < H && ix >= 0 && ix < W) {
                        const unsigned short* src =
                            X + (((size_t)bb * H + iy) * W + ix) * Cin + ci0 + ch * 8;
                        v = *(const uint4*)src;                 // global_load_b128
                        __builtin_prefetch(src + 32, 0, 0);     // global_prefetch_b8
                    }
                }
                // chunk ch -> fragment half/j-base; contiguous 16B LDS store
                *(uint4*)&aA[r >> 4][(ch & 1) * 16 + (r & 15)][(ch >> 1) << 3] = v;
            }
        } else {
            // generic scalar im2col gather (stem), scattered into fragment order
            for (int e = tid; e < 4096; e += 256) {
                int r = e >> 5, kk = e & 31;
                int m = mBase + r, k = k0 + kk;
                unsigned short v = 0;
                if (m < M && k < K) {
                    int ci = k % Cin;
                    int t  = k / Cin;
                    int kx = t % KW;
                    int ky = t / KW;
                    int bb = m / HoWo;
                    int p  = m - bb * HoWo;
                    int oy = p / Wo;
                    int ox = p - oy * Wo;
                    int iy = oy * stride - pad + ky;
                    int ix = ox * stride - pad + kx;
                    if (iy >= 0 && iy < H && ix >= 0 && ix < W)
                        v = X[(((size_t)bb * H + iy) * W + ix) * Cin + ci];
                }
                aA[r >> 4][((kk >> 3) & 1) * 16 + (r & 15)][(kk & 7) + ((kk >> 4) << 3)] = v;
            }
        }
        __syncthreads();

        // B fragments straight from packed global weights (two 32B per lane)
        v16bf bv[2];
#pragma unroll
        for (int sn = 0; sn < 2; sn++) {
            size_t o = ((size_t)kb * NT + (ntBase + wn * 2 + sn)) * 512 + (size_t)lane * 16;
            bv[sn] = *(const v16bf*)(Wt + o);
        }
        // A fragments: one contiguous 32B LDS read per lane per subtile
        v16bf av[2];
#pragma unroll
        for (int sm = 0; sm < 2; sm++)
            av[sm] = *(const v16bf*)&aA[wm * 2 + sm][lane][0];

#pragma unroll
        for (int sm = 0; sm < 2; sm++)
#pragma unroll
            for (int sn = 0; sn < 2; sn++)
                acc[sm][sn] = __builtin_amdgcn_wmma_f32_16x16x32_bf16(
                    false, av[sm], false, bv[sn], (short)0, acc[sm][sn], false, false);
        __syncthreads();
    }

    // ---- epilogue: BN + residual + ReLU. C/D layout: col = lane&15, row = lh*8+r ----
#pragma unroll
    for (int sm = 0; sm < 2; sm++) {
#pragma unroll
        for (int sn = 0; sn < 2; sn++) {
            int n = (ntBase + wn * 2 + sn) * 16 + lm;
            if (n >= Cout) continue;
            float sc = scale ? scale[n] : 1.0f;
            float bi = scale ? bias[n]  : 0.0f;
#pragma unroll
            for (int r = 0; r < 8; r++) {
                int m = mBase + wm * 32 + sm * 16 + lh * 8 + r;
                if (m >= M) continue;
                float v = acc[sm][sn][r] * sc + bi;
                size_t o = (size_t)m * Cout + n;
                if (Res)  v += bf2f(Res[o]);
                if (relu) v = v > 0.0f ? v : 0.0f;
                if (Y)  Y[o]  = f2bf(v);
                if (Yf) Yf[o] = v;
            }
        }
    }
}

// ---------------------------------------------------------------------------
// Tail (analytic Jacobian, no autodiff):
//   z = inv*feat + bias + idn ; pooled = mean(relu(z)) ; logits = pooled^3 @ W^T
//   coef[b,c] = 3*pooled^2*inv/196 ; cam[b,n,p] = sum_c W[n,c]*coef*mask*feat
// ---------------------------------------------------------------------------
__global__ void k_tail_stats(const float* __restrict__ feat,          // f32 NHWC [8,196,2048]
                             const unsigned short* __restrict__ idn,  // bf16 NHWC
                             const float* __restrict__ scale3, const float* __restrict__ bias3,
                             const float* __restrict__ fcw,           // [20,2048]
                             float* __restrict__ probs,               // [8,20]
                             float* __restrict__ coef)                // [8,2048]
{
    int b = blockIdx.x, t = threadIdx.x;
    __shared__ float pooled[2048];
    __shared__ float lg[20];
    for (int c = t; c < 2048; c += 256) {
        float inv = scale3[c], bi = bias3[c], s = 0.0f;
        for (int p = 0; p < 196; p++) {
            size_t i = ((size_t)b * 196 + p) * 2048 + c;
            float z = feat[i] * inv + bi + bf2f(idn[i]);
            s += z > 0.0f ? z : 0.0f;
        }
        float pc = s * (1.0f / 196.0f);
        pooled[c] = pc;
        coef[(size_t)b * 2048 + c] = 3.0f * pc * pc * inv * (1.0f / 196.0f);
    }
    __syncthreads();
    if (t < 20) {
        float s = 0.0f;
        for (int c = 0; c < 2048; c++) {
            float p = pooled[c];
            s += p * p * p * fcw[(size_t)t * 2048 + c];
        }
        lg[t] = s;
    }
    __syncthreads();
    if (t == 0) {
        float mx = lg[0];
        for (int n = 1; n < 20; n++) mx = fmaxf(mx, lg[n]);
        float e[20], se = 0.0f;
        for (int n = 0; n < 20; n++) { e[n] = __expf(lg[n] - mx); se += e[n]; }
        for (int n = 0; n < 20; n++) probs[b * 20 + n] = e[n] / se;
    }
}

__global__ void k_tail_cam(const float* __restrict__ feat, const unsigned short* __restrict__ idn,
                           const float* __restrict__ scale3, const float* __restrict__ bias3,
                           const float* __restrict__ fcw, const float* __restrict__ coef,
                           float* __restrict__ cams)                  // [8,20,196]
{
    int bp = blockIdx.x;
    int b = bp / 196, p = bp % 196, t = threadIdx.x;
    float acc[20];
#pragma unroll
    for (int n = 0; n < 20; n++) acc[n] = 0.0f;
    for (int c = t; c < 2048; c += 256) {
        size_t i = ((size_t)b * 196 + p) * 2048 + c;
        float f = feat[i];
        float z = f * scale3[c] + bias3[c] + bf2f(idn[i]);
        if (z > 0.0f) {
            float v = coef[(size_t)b * 2048 + c] * f;
#pragma unroll
            for (int n = 0; n < 20; n++) acc[n] += fcw[(size_t)n * 2048 + c] * v;
        }
    }
    __shared__ float red[256][20];
    for (int n = 0; n < 20; n++) red[t][n] = acc[n];
    __syncthreads();
    for (int s = 128; s > 0; s >>= 1) {
        if (t < s)
            for (int n = 0; n < 20; n++) red[t][n] += red[t + s][n];
        __syncthreads();
    }
    if (t < 20) cams[((size_t)b * 20 + t) * 196 + p] = red[0][t];
}

// ---------------------------------------------------------------------------
// Host orchestration
// ---------------------------------------------------------------------------
struct Spec { int in_c, mid, stride, k2, down, dstride; };

static void make_specs(Spec* s) {
    int lm[4] = {64, 128, 256, 512}, ln[4] = {3, 4, 6, 3}, ls[4] = {1, 2, 2, 2};
    int in_c = 64, idx = 0;
    for (int li = 0; li < 4; li++)
        for (int bi = 0; bi < ln[li]; bi++) {
            int stride, k2, dstride;
            if (li == 3 && bi == 0) { stride = 1; k2 = 1; dstride = 1; }
            else { stride = (bi == 0) ? ls[li] : 1; k2 = 3; dstride = ls[li]; }
            s[idx].in_c = in_c; s[idx].mid = lm[li]; s[idx].stride = stride;
            s[idx].k2 = k2; s[idx].down = (bi == 0); s[idx].dstride = dstride;
            idx++;
            in_c = lm[li] * 4;
        }
}

struct Net {
    const float *x, *stem_w, *stem_bn[4], *fc_w;
    struct Blk { const float *w1, *bn1[4], *w2, *bn2[4], *w3, *bn3[4], *dw, *dbn[4]; } blk[16];
};

// mode 0: python dict insertion order. mode 1: jax pytree (sorted-key) order.
static int build_slots(int mode, Net* n, const float*** dst, long* sz, const Spec* sp) {
    int c = 0;
    auto add = [&](const float** d, long s) { dst[c] = d; sz[c] = s; c++; };
    if (mode == 0) {
        add(&n->x, 8L * 3 * 224 * 224);
        add(&n->stem_w, 64L * 3 * 7 * 7);
        for (int j = 0; j < 4; j++) add(&n->stem_bn[j], 64);
        for (int i = 0; i < 16; i++) {
            const Spec& s = sp[i];
            Net::Blk& b = n->blk[i];
            add(&b.w1, (long)s.mid * s.in_c);
            for (int j = 0; j < 4; j++) add(&b.bn1[j], s.mid);
            add(&b.w2, (long)s.mid * s.mid * s.k2 * s.k2);
            for (int j = 0; j < 4; j++) add(&b.bn2[j], s.mid);
            add(&b.w3, (long)s.mid * 4 * s.mid);
            for (int j = 0; j < 4; j++) add(&b.bn3[j], s.mid * 4);
            if (s.down) {
                add(&b.dw, (long)s.mid * 4 * s.in_c);
                for (int j = 0; j < 4; j++) add(&b.dbn[j], s.mid * 4);
            }
        }
        add(&n->fc_w, 20L * 2048);
    } else {
        for (int i = 0; i < 16; i++) {
            const Spec& s = sp[i];
            Net::Blk& b = n->blk[i];
            for (int j = 0; j < 4; j++) add(&b.bn1[j], s.mid);
            for (int j = 0; j < 4; j++) add(&b.bn2[j], s.mid);
            for (int j = 0; j < 4; j++) add(&b.bn3[j], s.mid * 4);
            if (s.down) {
                for (int j = 0; j < 4; j++) add(&b.dbn[j], s.mid * 4);
                add(&b.dw, (long)s.mid * 4 * s.in_c);
            }
            add(&b.w1, (long)s.mid * s.in_c);
            add(&b.w2, (long)s.mid * s.mid * s.k2 * s.k2);
            add(&b.w3, (long)s.mid * 4 * s.mid);
        }
        add(&n->fc_w, 20L * 2048);
        for (int j = 0; j < 4; j++) add(&n->stem_bn[j], 64);
        add(&n->stem_w, 64L * 3 * 7 * 7);
        add(&n->x, 8L * 3 * 224 * 224);
    }
    return c;
}

static inline long cdiv(long a, long b) { return (a + b - 1) / b; }

static void conv_launch(hipStream_t st, const unsigned short* X, const unsigned short* Wt,
                        const float* scale, const float* bias, const unsigned short* Res,
                        unsigned short* Y, float* Yf,
                        int Bn, int H, int W, int Cin, int Cout,
                        int KH, int KW, int stride, int pad, int Ho, int Wo, int relu) {
    int M = Bn * Ho * Wo;
    dim3 g((unsigned)cdiv(M, 128), (unsigned)(Cout / 64));
    k_conv_wmma<<<g, 256, 0, st>>>(X, Wt, scale, bias, Res, Y, Yf,
                                   Bn, H, W, Cin, Cout, KH, KW, stride, pad, Ho, Wo, relu);
}

extern "C" void kernel_launch(void* const* d_in, const int* in_sizes, int n_in,
                              void* d_out, int out_size, void* d_ws, size_t ws_size,
                              hipStream_t stream)
{
    (void)out_size; (void)ws_size;
    Spec sp[16];
    make_specs(sp);

    // ---- resolve input flattening order via in_sizes fingerprint ----
    Net net;
    const float** dst[300];
    long sz[300];
    int cnt = build_slots(0, &net, dst, sz, sp);
    bool ok = (cnt == n_in);
    if (ok) for (int i = 0; i < cnt; i++) if ((long)in_sizes[i] != sz[i]) { ok = false; break; }
    if (!ok) {
        int c1 = build_slots(1, &net, dst, sz, sp);
        bool ok1 = (c1 == n_in);
        if (ok1) for (int i = 0; i < c1; i++) if ((long)in_sizes[i] != sz[i]) { ok1 = false; break; }
        if (ok1) cnt = c1;
        else     cnt = build_slots(0, &net, dst, sz, sp);   // fallback
    }
    for (int i = 0; i < cnt && i < n_in; i++) *dst[i] = (const float*)d_in[i];

    // ---- workspace bump allocator ----
    char* wsp = (char*)d_ws;
    size_t off = 0;
    auto alloc = [&](size_t bytes) -> void* {
        void* p = wsp + off;
        off += (bytes + 255) & ~(size_t)255;
        return p;
    };
    unsigned short* xin = (unsigned short*)alloc((size_t)8 * 224 * 224 * 3 * 2);
    unsigned short* P0  = (unsigned short*)alloc((size_t)6422528 * 2);
    unsigned short* P1  = (unsigned short*)alloc((size_t)6422528 * 2);
    unsigned short* T1  = (unsigned short*)alloc((size_t)3211264 * 2);
    unsigned short* T2  = (unsigned short*)alloc((size_t)1605632 * 2);
    unsigned short* DSb = (unsigned short*)alloc((size_t)6422528 * 2);
    float* featF = (float*)alloc((size_t)3211264 * 4);   // [8,14,14,2048]
    float* coef  = (float*)alloc((size_t)8 * 2048 * 4);

    auto prep = [&](const float* w, int O, int I, int KH, int KW, const float* const* bn,
                    unsigned short** pw, float** psc, float** pbi) {
        int  K    = KH * KW * I;
        int  Kpad = (int)cdiv(K, 32) * 32;
        long total = (long)Kpad * O;
        *pw = (unsigned short*)alloc((size_t)total * 2);
        k_prep_w<<<(unsigned)cdiv(total, 256), 256, 0, stream>>>(w, *pw, O, I, KH, KW, Kpad);
        *psc = (float*)alloc((size_t)O * 4);
        *pbi = (float*)alloc((size_t)O * 4);
        k_prep_bn<<<(unsigned)cdiv(O, 256), 256, 0, stream>>>(bn[0], bn[1], bn[2], bn[3],
                                                              *psc, *pbi, O);
    };

    // ---- stem: conv7x7 s2 p3 + BN + ReLU, then 3x3 s2 maxpool ----
    k_x_to_nhwc<<<(unsigned)cdiv(8L * 3 * 224 * 224, 256), 256, 0, stream>>>(net.x, xin, 8, 3, 224, 224);
    unsigned short* swt; float *ssc, *sbi;
    prep(net.stem_w, 64, 3, 7, 7, net.stem_bn, &swt, &ssc, &sbi);
    conv_launch(stream, xin, swt, ssc, sbi, nullptr, P0, nullptr,
                8, 224, 224, 3, 64, 7, 7, 2, 3, 112, 112, 1);
    k_maxpool<<<(unsigned)cdiv(8L * 56 * 56 * 64, 256), 256, 0, stream>>>(P0, P1, 8, 64, 112, 112, 56, 56);

    // ---- 16 bottleneck blocks ----
    unsigned short* cur = P1;
    unsigned short* alt = P0;
    int Hc = 56;
    float* sc3_last = nullptr; float* bi3_last = nullptr;
    for (int i = 0; i < 16; i++) {
        const Spec& s = sp[i];
        int mid = s.mid, inc = s.in_c, k2 = s.k2;
        int Ho = Hc / s.stride;
        unsigned short *w1, *w2, *w3, *dw;
        float *s1, *b1, *s2, *b2, *s3, *b3, *sd = nullptr, *bd = nullptr;
        prep(net.blk[i].w1, mid, inc, 1, 1, net.blk[i].bn1, &w1, &s1, &b1);
        prep(net.blk[i].w2, mid, mid, k2, k2, net.blk[i].bn2, &w2, &s2, &b2);
        prep(net.blk[i].w3, mid * 4, mid, 1, 1, net.blk[i].bn3, &w3, &s3, &b3);

        conv_launch(stream, cur, w1, s1, b1, nullptr, T1, nullptr,
                    8, Hc, Hc, inc, mid, 1, 1, 1, 0, Hc, Hc, 1);
        conv_launch(stream, T1, w2, s2, b2, nullptr, T2, nullptr,
                    8, Hc, Hc, mid, mid, k2, k2, s.stride, (k2 - 1) / 2, Ho, Ho, 1);

        const unsigned short* res = cur;
        if (s.down) {
            prep(net.blk[i].dw, mid * 4, inc, 1, 1, net.blk[i].dbn, &dw, &sd, &bd);
            conv_launch(stream, cur, dw, sd, bd, nullptr, DSb, nullptr,
                        8, Hc, Hc, inc, mid * 4, 1, 1, s.dstride, 0, Ho, Ho, 0);
            res = DSb;
        }
        if (i < 15) {
            // y = relu(bn3(conv3) + idn)
            conv_launch(stream, T2, w3, s3, b3, res, alt, nullptr,
                        8, Ho, Ho, mid, mid * 4, 1, 1, 1, 0, Ho, Ho, 1);
            unsigned short* t = cur; cur = alt; alt = t;
        } else {
            // hooked conv3: raw f32 feature map (no BN / res / relu); idn = cur
            conv_launch(stream, T2, w3, nullptr, nullptr, nullptr, nullptr, featF,
                        8, Ho, Ho, mid, mid * 4, 1, 1, 1, 0, Ho, Ho, 0);
            sc3_last = s3; bi3_last = b3;
        }
        Hc = Ho;
    }

    // ---- analytic tail: pooled/logits/softmax + closed-form Jacobian CAM ----
    float* outp = (float*)d_out;
    k_tail_stats<<<8, 256, 0, stream>>>(featF, cur, sc3_last, bi3_last, net.fc_w, outp, coef);
    k_tail_cam<<<8 * 196, 256, 0, stream>>>(featF, cur, sc3_last, bi3_last, net.fc_w, coef,
                                            outp + 160);
}
